// EMAQuantizer_10024453669334
// MI455X (gfx1250) — compile-verified, matrix operand order checked by reading, not run on hardware
//
#include <hip/hip_runtime.h>
#include <hip/hip_bf16.h>
#include <stdint.h>

// ---------------------------------------------------------------------------
// VQ-VAE EMA quantizer (eval path) for gfx1250.
//   d_out layout (floats): [quantized 16777216][loss 1][indices 131072]
//                          [encodings_sum 256][embedding 131072]
// ---------------------------------------------------------------------------

typedef __attribute__((ext_vector_type(16))) __bf16 v16bf;
typedef __attribute__((ext_vector_type(8)))  float  v8f;

#define C_DIM      128
#define K_CODES    1024
#define S_SPATIAL  32768               // D*H*W
#define N_VECS     131072              // B * S_SPATIAL
#define CHUNK      128                 // codebook rows staged in LDS at a time
#define NCHUNK     (K_CODES / CHUNK)   // 8
#define TILES      (CHUNK / 16)        // 8 code-tiles per chunk

__device__ __forceinline__ __bf16 f2bf(float f) {
    unsigned u = __builtin_bit_cast(unsigned, f);
    u += 0x7FFFu + ((u >> 16) & 1u);               // round-to-nearest-even
    unsigned short h = (unsigned short)(u >> 16);
    return __builtin_bit_cast(__bf16, h);
}

// ---------------------------------------------------------------------------
// Kernel 1: nearest-codebook search via bf16 WMMA.
// Each wave handles 16 consecutive vectors (rows); block = 8 waves = 128 rows.
// score[n,k] = x_n . e_k - 0.5*||e_k||^2   (argmax == argmin distance)
// ---------------------------------------------------------------------------
__global__ __launch_bounds__(256) void vq_argmin(const float* __restrict__ inp,
                                                 const float* __restrict__ emb,
                                                 float* __restrict__ idx_out)
{
    // codebook chunk pre-swizzled into WMMA B-fragment layout:
    // [tile(8)][frag(4)][lane(32)][16 contiguous bf16]
    __shared__ __align__(32) __bf16 cb[TILES * 4 * 32 * 16];   // 32 KB
    __shared__ float hn[CHUNK];                                // 0.5*||e||^2

    const int tid  = threadIdx.x;
    const int lane = tid & 31;
    const int wave = tid >> 5;
    const int col  = lane & 15;        // matrix column / row-M within half
    const int kh   = lane >> 4;        // which K-half this lane holds

    const int n0 = blockIdx.x * 128 + wave * 16;          // first row of tile
    const int b  = n0 >> 15;                              // n0 / S_SPATIAL
    const int s0 = n0 & (S_SPATIAL - 1);
    const float* inB = inp + (size_t)b * (size_t)C_DIM * S_SPATIAL;

    // ---- load A fragments once (16 rows x 128 C, bf16, ISA 16-bit A layout)
    v16bf a[4];
#pragma unroll
    for (int f = 0; f < 4; ++f) {
#pragma unroll
        for (int e = 0; e < 16; ++e) {
            int c = f * 32 + kh * 8 + e + ((e >= 8) ? 8 : 0);
            a[f][e] = f2bf(inB[(size_t)c * S_SPATIAL + (s0 + col)]);
        }
    }

    float bestv[8];
    int   besti[8];
#pragma unroll
    for (int i = 0; i < 8; ++i) { bestv[i] = -3.4e38f; besti[i] = 0; }

    for (int ch = 0; ch < NCHUNK; ++ch) {
        __syncthreads();
        // ---- stage codebook chunk: fp32 global -> swizzled bf16 LDS
        const float* ec = emb + (size_t)ch * CHUNK * C_DIM;
        for (int j = tid; j < CHUNK * C_DIM; j += 256) {
            int code = j >> 7;             // / C_DIM
            int c    = j & (C_DIM - 1);
            int f    = c >> 5;
            int r    = c & 31;
            int kh2, e2;
            if (r < 16) { kh2 = r >> 3; e2 = r & 7; }
            else        { int r2 = r - 16; kh2 = r2 >> 3; e2 = 8 + (r2 & 7); }
            int ln = kh2 * 16 + (code & 15);
            int kt = code >> 4;
            cb[(((kt * 4 + f) * 32) + ln) * 16 + e2] = f2bf(ec[j]);
        }
        if (tid < CHUNK) {
            const float* er = ec + (size_t)tid * C_DIM;
            float s = 0.f;
#pragma unroll 8
            for (int c = 0; c < C_DIM; ++c) { float v = er[c]; s += v * v; }
            hn[tid] = 0.5f * s;
        }
        __syncthreads();

        // ---- 8 code-tiles per chunk; 4 WMMA (K=32) each
#pragma unroll
        for (int kt = 0; kt < TILES; ++kt) {
            v8f acc = {};
#pragma unroll
            for (int f = 0; f < 4; ++f) {
                v16bf bf = *(const v16bf*)&cb[(((kt * 4 + f) * 32) + lane) * 16];
                acc = __builtin_amdgcn_wmma_f32_16x16x32_bf16(
                          false, a[f], false, bf, (short)0, acc, false, false);
            }
            float h  = hn[kt * 16 + col];
            int   kc = ch * CHUNK + kt * 16 + col;
#pragma unroll
            for (int i = 0; i < 8; ++i) {
                float sc = acc[i] - h;
                if (sc > bestv[i]) { bestv[i] = sc; besti[i] = kc; }
            }
        }
    }

    // ---- argmax across the 16 columns (lanes 0..15 hold rows 0..7,
    //      lanes 16..31 hold rows 8..15; xor offsets stay within each half)
#pragma unroll
    for (int i = 0; i < 8; ++i) {
#pragma unroll
        for (int off = 8; off > 0; off >>= 1) {
            float ov = __shfl_xor(bestv[i], off, 32);
            int   oi = __shfl_xor(besti[i], off, 32);
            if (ov > bestv[i] || (ov == bestv[i] && oi < besti[i])) {
                bestv[i] = ov; besti[i] = oi;
            }
        }
    }
    if (col == 0) {
        int rowbase = n0 + kh * 8;
#pragma unroll
        for (int i = 0; i < 8; ++i)
            idx_out[rowbase + i] = (float)besti[i];
    }
}

// ---------------------------------------------------------------------------
// Kernel 2: gather fp32 codebook rows -> quantized (channel-first), and
// accumulate commitment loss = 2.5 * mean((q - x)^2).
// ---------------------------------------------------------------------------
__global__ __launch_bounds__(256) void vq_gather_loss(const float* __restrict__ inp,
                                                      const float* __restrict__ emb,
                                                      const float* __restrict__ idxf,
                                                      float* __restrict__ outq,
                                                      float* __restrict__ loss)
{
    const int n  = blockIdx.x * 256 + threadIdx.x;
    const int idx = (int)(idxf[n] + 0.5f);
    const int b  = n >> 15;
    const int s  = n & (S_SPATIAL - 1);
    const float4* ev = (const float4*)(emb + (size_t)idx * C_DIM);
    const size_t base = (size_t)b * (size_t)C_DIM * S_SPATIAL + s;

    float acc = 0.f;
#pragma unroll 8
    for (int c4 = 0; c4 < C_DIM / 4; ++c4) {
        float4 e = ev[c4];
        size_t p = base + (size_t)(c4 * 4) * S_SPATIAL;
        float x0 = inp[p];
        float x1 = inp[p + 1 * (size_t)S_SPATIAL];
        float x2 = inp[p + 2 * (size_t)S_SPATIAL];
        float x3 = inp[p + 3 * (size_t)S_SPATIAL];
        outq[p]                           = e.x;
        outq[p + 1 * (size_t)S_SPATIAL]   = e.y;
        outq[p + 2 * (size_t)S_SPATIAL]   = e.z;
        outq[p + 3 * (size_t)S_SPATIAL]   = e.w;
        float d0 = e.x - x0, d1 = e.y - x1, d2 = e.z - x2, d3 = e.w - x3;
        acc += d0 * d0 + d1 * d1 + d2 * d2 + d3 * d3;
    }

    __shared__ float red[256];
    red[threadIdx.x] = acc;
    __syncthreads();
    for (int st = 128; st > 0; st >>= 1) {
        if (threadIdx.x < st) red[threadIdx.x] += red[threadIdx.x + st];
        __syncthreads();
    }
    if (threadIdx.x == 0)
        atomicAdd(loss, red[0] * (2.5f / 16777216.f));
}

// ---------------------------------------------------------------------------
extern "C" void kernel_launch(void* const* d_in, const int* in_sizes, int n_in,
                              void* d_out, int out_size, void* d_ws, size_t ws_size,
                              hipStream_t stream)
{
    const float* inp = (const float*)d_in[0];   // [4,128,32,32,32]
    const float* emb = (const float*)d_in[1];   // [1024,128]
    float* out = (float*)d_out;

    const size_t Q = (size_t)4 * C_DIM * S_SPATIAL;   // 16777216
    float* loss   = out + Q;
    float* idxf   = out + Q + 1;
    float* encsum = out + Q + 1 + N_VECS;
    float* embout = encsum + 256;

    // zero loss accumulator + encodings_sum (graph-capturable)
    hipMemsetAsync(loss,   0, sizeof(float),         stream);
    hipMemsetAsync(encsum, 0, 256 * sizeof(float),   stream);

    vq_argmin<<<N_VECS / 128, 256, 0, stream>>>(inp, emb, idxf);
    vq_gather_loss<<<N_VECS / 256, 256, 0, stream>>>(inp, emb, idxf, out, loss);

    // pass-through embedding output
    hipMemcpyAsync(embout, emb, (size_t)K_CODES * C_DIM * sizeof(float),
                   hipMemcpyDeviceToDevice, stream);
}
